// QRNNCell_64115271795099
// MI455X (gfx1250) — compile-verified, matrix operand order checked by reading, not run on hardware
//
#include <hip/hip_runtime.h>
#include <hip/hip_bf16.h>

// Problem constants (from reference): B=64, D=512.
#define DD      512
#define BTOT    64
#define BLK_M   64
#define BLK_N   64
#define KSTEP   32
#define LDK     40      // padded LDS row stride in bf16 elems (80B: keeps 8/16B alignment, spreads banks)
#define BCHUNK  16      // batches per workspace chunk -> ws = 2*16*512*512*4 = 32 MB

typedef __attribute__((ext_vector_type(16))) __bf16 v16bf;
typedef __attribute__((ext_vector_type(8)))  __bf16 v8bf;
typedef __attribute__((ext_vector_type(4)))  __bf16 v4bf;
typedef __attribute__((ext_vector_type(8)))  float  v8f;
typedef __attribute__((ext_vector_type(4)))  float  v4f;

// Split f32 into bf16 hi + bf16 lo (error term). hi+lo carries ~16 mantissa bits.
__device__ __forceinline__ void split4(v4f a, v4bf& h, v4bf& l) {
#pragma unroll
    for (int j = 0; j < 4; ++j) {
        __bf16 hh = (__bf16)a[j];
        h[j] = hh;
        l[j] = (__bf16)(a[j] - (float)hh);
    }
}

__device__ __forceinline__ void split8(const float* a, v8bf& h, v8bf& l) {
#pragma unroll
    for (int j = 0; j < 8; ++j) {
        __bf16 hh = (__bf16)a[j];
        h[j] = hh;
        l[j] = (__bf16)(a[j] - (float)hh);
    }
}

// Load a 16x32 bf16 WMMA operand fragment for this lane from an LDS row
// laid out [row][K] with stride LDK. Per ISA 7.12.2 (16-bit A, wave32):
//   lanes 0-15 : row = lane,    frag elems 0..7 = K 0..7,  8..15 = K 16..23
//   lanes 16-31: row = lane-16, frag elems 0..7 = K 8..15, 8..15 = K 24..31
// Caller passes p = &lds[row*LDK + kb] with kb = (lane>=16)*8. B operand uses
// the mirrored layout from K-major storage (column index plays the row role).
__device__ __forceinline__ v16bf ldfrag(const __bf16* p) {
    union { v16bf v; v8bf h[2]; } u;
    u.h[0] = *(const v8bf*)(p);
    u.h[1] = *(const v8bf*)(p + 16);
    return u.v;
}

__device__ __forceinline__ v8f wmma_bf16(v16bf a, v16bf b, v8f c) {
    return __builtin_amdgcn_wmma_f32_16x16x32_bf16(false, a, false, b, (short)0, c,
                                                   false, false);
}

// ---------------------------------------------------------------------------
// Stage 1:  T = U @ X (complex), per batch.  Tr = Ur*Xr - Ui*Xi, Ti = Ui*Xr + Ur*Xi
// A = U rows (row-major, K contiguous).  B = X (K-major needed -> transposed LDS store).
// ---------------------------------------------------------------------------
__global__ __launch_bounds__(256) void evo_stage1(
    const float* __restrict__ Ur, const float* __restrict__ Ui,
    const float* __restrict__ Xr, const float* __restrict__ Xi,
    float* __restrict__ Tr, float* __restrict__ Ti, int b0)
{
    __shared__ alignas(16) __bf16 sA[4][BLK_M * LDK]; // Ur_h, Ur_l, Ui_h, Ui_l : [m][k]
    __shared__ alignas(16) __bf16 sB[4][BLK_N * LDK]; // Xr_h, Xr_l, Xi_h, Xi_l : [n][k]

    const int bm = blockIdx.x * BLK_M;
    const int bn = blockIdx.y * BLK_N;
    const int zb = blockIdx.z;          // chunk-local batch
    const int b  = b0 + zb;             // global batch

    const float* xr = Xr + (size_t)b * DD * DD;
    const float* xi = Xi + (size_t)b * DD * DD;
    float* tr = Tr + (size_t)zb * DD * DD;
    float* ti = Ti + (size_t)zb * DD * DD;

    const int tid  = threadIdx.x;
    const int lane = tid & 31;
    const int wave = tid >> 5;
    const int wm   = wave & 1;          // 2 waves over M (32 rows each)
    const int wn   = wave >> 1;         // 4 waves over N (16 cols each)
    const int half = lane >> 4;
    const int kb   = half * 8;
    const int l15  = lane & 15;

    // B-stage load mapping: thread owns column n and 8 consecutive q values.
    const int bln = tid & 63;           // n within tile
    const int blq = (tid >> 6) * 8;     // q0: 0,8,16,24

    v8f accRR[2] = {}; // Ur*Xr
    v8f accII[2] = {}; // Ui*Xi
    v8f accTI[2] = {}; // Ui*Xr + Ur*Xi

    for (int kt = 0; kt < DD; kt += KSTEP) {
        __syncthreads();
        // --- A tiles: U[bm..bm+64) x [kt..kt+32): float4 loads, b64 packed stores ---
#pragma unroll
        for (int i = 0; i < 2; ++i) {
            int s  = tid + i * 256;             // 0..511 float4 slots
            int r  = s >> 3;
            int c4 = (s & 7) * 4;
            v4f a0 = *(const v4f*)&Ur[(size_t)(bm + r) * DD + kt + c4];
            v4f a1 = *(const v4f*)&Ui[(size_t)(bm + r) * DD + kt + c4];
            v4bf h, l;
            split4(a0, h, l);
            *(v4bf*)&sA[0][r * LDK + c4] = h;
            *(v4bf*)&sA[1][r * LDK + c4] = l;
            split4(a1, h, l);
            *(v4bf*)&sA[2][r * LDK + c4] = h;
            *(v4bf*)&sA[3][r * LDK + c4] = l;
        }
        // --- B tiles: X[kt..kt+32) x [bn..bn+64), stored transposed [n][k].
        //     8 coalesced b32 loads down column n -> contiguous in LDS -> b128 stores.
        {
            float vr[8], vi[8];
#pragma unroll
            for (int j = 0; j < 8; ++j) {
                vr[j] = xr[(size_t)(kt + blq + j) * DD + bn + bln];
                vi[j] = xi[(size_t)(kt + blq + j) * DD + bn + bln];
            }
            v8bf h, l;
            split8(vr, h, l);
            *(v8bf*)&sB[0][bln * LDK + blq] = h;
            *(v8bf*)&sB[1][bln * LDK + blq] = l;
            split8(vi, h, l);
            *(v8bf*)&sB[2][bln * LDK + blq] = h;
            *(v8bf*)&sB[3][bln * LDK + blq] = l;
        }
        __syncthreads();

        const int nrow = (wn * 16 + l15) * LDK + kb;
        v16bf Xrh = ldfrag(&sB[0][nrow]);
        v16bf Xrl = ldfrag(&sB[1][nrow]);
        v16bf Xih = ldfrag(&sB[2][nrow]);
        v16bf Xil = ldfrag(&sB[3][nrow]);

#pragma unroll
        for (int mt = 0; mt < 2; ++mt) {
            const int ar = (wm * 32 + mt * 16 + l15) * LDK + kb;
            v16bf Urh = ldfrag(&sA[0][ar]);
            v16bf Url = ldfrag(&sA[1][ar]);
            v16bf Uih = ldfrag(&sA[2][ar]);
            v16bf Uil = ldfrag(&sA[3][ar]);

            // Ur*Xr (split: hh + hl + lh)
            accRR[mt] = wmma_bf16(Urh, Xrh, accRR[mt]);
            accRR[mt] = wmma_bf16(Urh, Xrl, accRR[mt]);
            accRR[mt] = wmma_bf16(Url, Xrh, accRR[mt]);
            // Ui*Xi
            accII[mt] = wmma_bf16(Uih, Xih, accII[mt]);
            accII[mt] = wmma_bf16(Uih, Xil, accII[mt]);
            accII[mt] = wmma_bf16(Uil, Xih, accII[mt]);
            // Ti = Ui*Xr + Ur*Xi (same sign -> shared accumulator)
            accTI[mt] = wmma_bf16(Uih, Xrh, accTI[mt]);
            accTI[mt] = wmma_bf16(Uih, Xrl, accTI[mt]);
            accTI[mt] = wmma_bf16(Uil, Xrh, accTI[mt]);
            accTI[mt] = wmma_bf16(Urh, Xih, accTI[mt]);
            accTI[mt] = wmma_bf16(Urh, Xil, accTI[mt]);
            accTI[mt] = wmma_bf16(Url, Xih, accTI[mt]);
        }
    }

    // C/D layout (ISA 7.12.2): VGPR v -> M = v + 8*half, N = l15.
#pragma unroll
    for (int mt = 0; mt < 2; ++mt) {
        const int row0 = bm + wm * 32 + mt * 16 + half * 8;
        const int col  = bn + wn * 16 + l15;
#pragma unroll
        for (int v = 0; v < 8; ++v) {
            tr[(size_t)(row0 + v) * DD + col] = accRR[mt][v] - accII[mt][v];
            ti[(size_t)(row0 + v) * DD + col] = accTI[mt][v];
        }
    }
}

// ---------------------------------------------------------------------------
// Stage 2:  Out = T @ U^H, gated.
//   Or[p,q] = sum_k Tr[p,k]*Ur[q,k] + Ti[p,k]*Ui[q,k]
//   Oi[p,q] = sum_k Ti[p,k]*Ur[q,k] - Tr[p,k]*Ui[q,k]
// Both operands are K-contiguous rows ("NT" GEMM) -> no transpose anywhere.
// accum==0: out = g*val   (x phase, initializes d_out)
// accum==1: out += (1-g)*val (h phase)
// ---------------------------------------------------------------------------
__global__ __launch_bounds__(256) void evo_stage2(
    const float* __restrict__ Tr, const float* __restrict__ Ti,
    const float* __restrict__ Ur, const float* __restrict__ Ui,
    const float* __restrict__ Lambda,
    float* __restrict__ out_r, float* __restrict__ out_i,
    int b0, int accum)
{
    __shared__ alignas(16) __bf16 sA[4][BLK_M * LDK]; // Tr_h, Tr_l, Ti_h, Ti_l : [p][k]
    __shared__ alignas(16) __bf16 sB[4][BLK_N * LDK]; // Ur_h, Ur_l, Ui_h, Ui_l : [q][k]

    const int bm = blockIdx.x * BLK_M;
    const int bn = blockIdx.y * BLK_N;
    const int zb = blockIdx.z;
    const int b  = b0 + zb;

    const float* trp = Tr + (size_t)zb * DD * DD;
    const float* tip = Ti + (size_t)zb * DD * DD;

    const int tid  = threadIdx.x;
    const int lane = tid & 31;
    const int wave = tid >> 5;
    const int wm   = wave & 1;
    const int wn   = wave >> 1;
    const int half = lane >> 4;
    const int kb   = half * 8;
    const int l15  = lane & 15;

    v8f accOR[2] = {}; // Tr*Ur + Ti*Ui
    v8f accI1[2] = {}; // Ti*Ur
    v8f accI2[2] = {}; // Tr*Ui

    for (int kt = 0; kt < DD; kt += KSTEP) {
        __syncthreads();
        // Both tiles row-major: float4 loads, packed b64 stores per split plane.
#pragma unroll
        for (int i = 0; i < 2; ++i) {
            int s  = tid + i * 256;
            int r  = s >> 3;
            int c4 = (s & 7) * 4;
            v4bf h, l;
            v4f a0 = *(const v4f*)&trp[(size_t)(bm + r) * DD + kt + c4];
            v4f a1 = *(const v4f*)&tip[(size_t)(bm + r) * DD + kt + c4];
            split4(a0, h, l);
            *(v4bf*)&sA[0][r * LDK + c4] = h;
            *(v4bf*)&sA[1][r * LDK + c4] = l;
            split4(a1, h, l);
            *(v4bf*)&sA[2][r * LDK + c4] = h;
            *(v4bf*)&sA[3][r * LDK + c4] = l;

            v4f u0 = *(const v4f*)&Ur[(size_t)(bn + r) * DD + kt + c4];
            v4f u1 = *(const v4f*)&Ui[(size_t)(bn + r) * DD + kt + c4];
            split4(u0, h, l);
            *(v4bf*)&sB[0][r * LDK + c4] = h;
            *(v4bf*)&sB[1][r * LDK + c4] = l;
            split4(u1, h, l);
            *(v4bf*)&sB[2][r * LDK + c4] = h;
            *(v4bf*)&sB[3][r * LDK + c4] = l;
        }
        __syncthreads();

        const int nrow = (wn * 16 + l15) * LDK + kb;
        v16bf Urh = ldfrag(&sB[0][nrow]);
        v16bf Url = ldfrag(&sB[1][nrow]);
        v16bf Uih = ldfrag(&sB[2][nrow]);
        v16bf Uil = ldfrag(&sB[3][nrow]);

#pragma unroll
        for (int mt = 0; mt < 2; ++mt) {
            const int ar = (wm * 32 + mt * 16 + l15) * LDK + kb;
            v16bf Trh = ldfrag(&sA[0][ar]);
            v16bf Trl = ldfrag(&sA[1][ar]);
            v16bf Tih = ldfrag(&sA[2][ar]);
            v16bf Til = ldfrag(&sA[3][ar]);

            // Or = Tr*Ur + Ti*Ui (shared accumulator)
            accOR[mt] = wmma_bf16(Trh, Urh, accOR[mt]);
            accOR[mt] = wmma_bf16(Trh, Url, accOR[mt]);
            accOR[mt] = wmma_bf16(Trl, Urh, accOR[mt]);
            accOR[mt] = wmma_bf16(Tih, Uih, accOR[mt]);
            accOR[mt] = wmma_bf16(Tih, Uil, accOR[mt]);
            accOR[mt] = wmma_bf16(Til, Uih, accOR[mt]);
            // Oi = Ti*Ur - Tr*Ui
            accI1[mt] = wmma_bf16(Tih, Urh, accI1[mt]);
            accI1[mt] = wmma_bf16(Tih, Url, accI1[mt]);
            accI1[mt] = wmma_bf16(Til, Urh, accI1[mt]);
            accI2[mt] = wmma_bf16(Trh, Uih, accI2[mt]);
            accI2[mt] = wmma_bf16(Trh, Uil, accI2[mt]);
            accI2[mt] = wmma_bf16(Trl, Uih, accI2[mt]);
        }
    }

    const float lam = Lambda[0];
    const float g = 1.0f / (1.0f + __expf(-lam));
    const float w = accum ? (1.0f - g) : g;

#pragma unroll
    for (int mt = 0; mt < 2; ++mt) {
        const int row0 = bm + wm * 32 + mt * 16 + half * 8;
        const int col  = bn + wn * 16 + l15;
#pragma unroll
        for (int v = 0; v < 8; ++v) {
            size_t idx = (size_t)b * DD * DD + (size_t)(row0 + v) * DD + col;
            float Or = accOR[mt][v];
            float Oi = accI1[mt][v] - accI2[mt][v];
            if (accum) {
                out_r[idx] += w * Or;
                out_i[idx] += w * Oi;
            } else {
                out_r[idx] = w * Or;
                out_i[idx] = w * Oi;
            }
        }
    }
}

extern "C" void kernel_launch(void* const* d_in, const int* in_sizes, int n_in,
                              void* d_out, int out_size, void* d_ws, size_t ws_size,
                              hipStream_t stream) {
    const float* xr  = (const float*)d_in[0];
    const float* xi  = (const float*)d_in[1];
    const float* hr  = (const float*)d_in[2];
    const float* hi  = (const float*)d_in[3];
    const float* Uxr = (const float*)d_in[4];
    const float* Uxi = (const float*)d_in[5];
    const float* Uhr = (const float*)d_in[6];
    const float* Uhi = (const float*)d_in[7];
    const float* Lam = (const float*)d_in[8];

    float* out_r = (float*)d_out;
    float* out_i = out_r + (size_t)BTOT * DD * DD;

    float* wsTr = (float*)d_ws;                       // BCHUNK*D*D f32
    float* wsTi = wsTr + (size_t)BCHUNK * DD * DD;    // BCHUNK*D*D f32

    dim3 blk(256, 1, 1);
    dim3 grd(DD / BLK_M, DD / BLK_N, BCHUNK);
    for (int b0 = 0; b0 < BTOT; b0 += BCHUNK) {
        evo_stage1<<<grd, blk, 0, stream>>>(Uxr, Uxi, xr, xi, wsTr, wsTi, b0);
        evo_stage2<<<grd, blk, 0, stream>>>(wsTr, wsTi, Uxr, Uxi, Lam, out_r, out_i, b0, 0);
        evo_stage1<<<grd, blk, 0, stream>>>(Uhr, Uhi, hr, hi, wsTr, wsTi, b0);
        evo_stage2<<<grd, blk, 0, stream>>>(wsTr, wsTi, Uhr, Uhi, Lam, out_r, out_i, b0, 1);
    }
    (void)in_sizes; (void)n_in; (void)out_size; (void)ws_size;
}
